// STGCN_39805756900099
// MI455X (gfx1250) — matrix operand order repaired and checked by reading, not compiled, not run on hardware
//
#include <hip/hip_runtime.h>
#include <hip/hip_bf16.h>

#define N_NODES   50000
#define N_EDGES   800000
#define DIM       128
#define NUM_GRAPHS 64
#define NUM_CLASSES 3
#define ROWS_PER_BLK 80            // 50000 = 625 * 80 (exact)
#define LPAD (DIM + 8)             // 136 bf16 = 272 B row stride: 16B-aligned, bank-shift 4

typedef __attribute__((ext_vector_type(16))) __bf16 v16bf;
typedef __attribute__((ext_vector_type(8)))  __bf16 v8bf;
typedef __attribute__((ext_vector_type(8)))  float  v8f;

// ---------------- workspace layout (float offsets) ----------------
#define OFF_DINV 0                       // 50000 (padded to 50048)
#define OFF_NORM 50048                   // 800000
#define OFF_HA   850048                  // 6400000 (GEMM output)
#define OFF_HB   7250048                 // 6400000 (layer activations)
#define OFF_POOL 13650048                // 64*128
#define OFF_CNT  (OFF_POOL + NUM_GRAPHS*DIM)  // 64 (contiguous after pool)
#define OFF_G1   (OFF_CNT + NUM_GRAPHS)       // 64*128

// ---------------- degree / normalization ----------------
__global__ void init_deg(float* deg) {
  int i = blockIdx.x * blockDim.x + threadIdx.x;
  if (i < N_NODES) deg[i] = 1.0f;          // self-loop contributes 1
}

__global__ void deg_scatter(const long long* __restrict__ dst, float* deg) {
  int e = blockIdx.x * blockDim.x + threadIdx.x;
  if (e < N_EDGES) atomicAdd(&deg[(int)dst[e]], 1.0f);
}

__global__ void deg_to_dinv(float* deg) {
  int i = blockIdx.x * blockDim.x + threadIdx.x;
  if (i < N_NODES) deg[i] = rsqrtf(deg[i]);   // deg >= 1 always
}

__global__ void edge_norm(const long long* __restrict__ src,
                          const long long* __restrict__ dst,
                          const float* __restrict__ dinv,
                          float* __restrict__ norm) {
  int e = blockIdx.x * blockDim.x + threadIdx.x;
  if (e < N_EDGES) norm[e] = dinv[(int)src[e]] * dinv[(int)dst[e]];
}

// ---------------- WMMA GEMM: H = X @ W  (f32 in, bf16 MMA, f32 acc) -------------
// Block: 256 threads = 8 waves, owns 80 rows. Wave w computes column tile w
// (16 cols) for 5 row tiles, reusing its B fragment across all 5 WMMAs per K-step.
__global__ __launch_bounds__(256)
void gemm_xw(const float* __restrict__ X, const float* __restrict__ W,
             float* __restrict__ H) {
  __shared__ __align__(16) __bf16 sWt[DIM][LPAD];          // W transposed: sWt[n][k]
  __shared__ __align__(16) __bf16 sX[ROWS_PER_BLK][LPAD];  // X strip: sX[r][k]
  const int tid  = threadIdx.x;
  const int row0 = blockIdx.x * ROWS_PER_BLK;

  // convert + transpose weights into LDS (16384 elems, 64/thread)
  for (int i = tid; i < DIM * DIM; i += 256) {
    int k = i >> 7, n = i & 127;
    sWt[n][k] = (__bf16)W[i];
  }
  // convert 80-row X strip (10240 elems, 40/thread)
  for (int i = tid; i < ROWS_PER_BLK * DIM; i += 256) {
    int r = i >> 7, c = i & 127;
    sX[r][c] = (__bf16)X[(row0 + r) * DIM + c];
  }
  // prefetch next block's X strip while we compute (global_prefetch_b8)
  if (blockIdx.x + 1 < gridDim.x)
    __builtin_prefetch(&X[(row0 + ROWS_PER_BLK) * DIM + tid * 4], 0, 0);
  __syncthreads();

  const int wave = tid >> 5;        // 8 waves -> 8 column tiles of 16
  const int lane = tid & 31;
  const int half = lane >> 4;       // lane group within wave
  const int mn   = lane & 15;       // row (A) / col (B,C/D) index
  const int col0 = wave * 16;

  v8f acc[5] = {};
  const __bf16* brow = &sWt[col0 + mn][0];
  for (int kt = 0; kt < 4; ++kt) {  // K = 128 in 4 steps of 32
    const int kb = kt * 32;
    // B 32x16: lane = col, half selects K +8 within each 16-K block
    v8bf bq0 = *(const v8bf*)(brow + kb + half * 8);
    v8bf bq1 = *(const v8bf*)(brow + kb + 16 + half * 8);
    v16bf b = __builtin_shufflevector(bq0, bq1, 0, 1, 2, 3, 4, 5, 6, 7,
                                      8, 9, 10, 11, 12, 13, 14, 15);
#pragma unroll
    for (int rt = 0; rt < 5; ++rt) {
      // A 16x32: lane = row, half selects K +8 within each 16-K block
      const __bf16* arow = &sX[rt * 16 + mn][0];
      v8bf aq0 = *(const v8bf*)(arow + kb + half * 8);
      v8bf aq1 = *(const v8bf*)(arow + kb + 16 + half * 8);
      v16bf a = __builtin_shufflevector(aq0, aq1, 0, 1, 2, 3, 4, 5, 6, 7,
                                        8, 9, 10, 11, 12, 13, 14, 15);
      acc[rt] = __builtin_amdgcn_wmma_f32_16x16x32_bf16(
          /*neg_a=*/false, a, /*neg_b=*/false, b,
          /*c_mod=*/(short)0, acc[rt], /*reuse_a=*/false, /*reuse_b=*/false);
    }
  }
  // C/D: VGPR r holds row (r + 8*half), col = mn
#pragma unroll
  for (int rt = 0; rt < 5; ++rt)
#pragma unroll
    for (int r = 0; r < 8; ++r)
      H[(row0 + rt * 16 + r + half * 8) * DIM + col0 + mn] = acc[rt][r];
}

// ---------------- aggregation ----------------
// out[i] = dinv[i]^2 * h[i]   (self-loop term, also initializes the buffer)
__global__ void self_loop_init(const float* __restrict__ h,
                               const float* __restrict__ dinv,
                               float* __restrict__ agg) {
  int i = blockIdx.x * blockDim.x + threadIdx.x;   // float4 granularity
  if (i >= N_NODES * (DIM / 4)) return;
  int node = i >> 5;
  float w = dinv[node]; w *= w;
  float4 v = ((const float4*)h)[i];
  float4 o = {v.x * w, v.y * w, v.z * w, v.w * w};
  ((float4*)agg)[i] = o;
}

// one wave per edge: 32 lanes x float4 = 128 features
__global__ __launch_bounds__(256)
void edge_scatter(const long long* __restrict__ src,
                  const long long* __restrict__ dst,
                  const float* __restrict__ norm,
                  const float* __restrict__ h,
                  float* __restrict__ agg) {
  int e = blockIdx.x * (blockDim.x >> 5) + (threadIdx.x >> 5);
  if (e >= N_EDGES) return;
  int lane = threadIdx.x & 31;
  int s = (int)src[e], d = (int)dst[e];
  float w = norm[e];
  float4 v = ((const float4*)(h + (long)s * DIM))[lane];
  float* ap = agg + (long)d * DIM + lane * 4;
  atomicAdd(ap + 0, v.x * w);
  atomicAdd(ap + 1, v.y * w);
  atomicAdd(ap + 2, v.z * w);
  atomicAdd(ap + 3, v.w * w);
}

__global__ void bias_act(float* __restrict__ agg, const float* __restrict__ b,
                         int do_relu) {
  int i = blockIdx.x * blockDim.x + threadIdx.x;
  if (i >= N_NODES * DIM) return;
  float v = agg[i] + b[i & 127];
  agg[i] = do_relu ? fmaxf(v, 0.0f) : v;
}

// ---------------- pooling + classifier ----------------
__global__ void zero_buf(float* p, int n) {
  int i = blockIdx.x * blockDim.x + threadIdx.x;
  if (i < n) p[i] = 0.0f;
}

__global__ void pool_count(const long long* __restrict__ batch, float* counts) {
  int i = blockIdx.x * blockDim.x + threadIdx.x;
  if (i < N_NODES) atomicAdd(&counts[(int)batch[i]], 1.0f);
}

__global__ void pool_sum(const long long* __restrict__ batch,
                         const float* __restrict__ h, float* pooled) {
  int i = blockIdx.x * blockDim.x + threadIdx.x;
  if (i >= N_NODES * DIM) return;
  int node = i >> 7, f = i & 127;
  atomicAdd(&pooled[(int)batch[node] * DIM + f], h[i]);
}

__global__ void pool_div(float* pooled, const float* counts) {
  int i = blockIdx.x * blockDim.x + threadIdx.x;
  if (i >= NUM_GRAPHS * DIM) return;
  pooled[i] /= fmaxf(counts[i >> 7], 1.0f);
}

__global__ void mlp1(const float* __restrict__ pooled,
                     const float* __restrict__ C1,
                     const float* __restrict__ bc1, float* __restrict__ g1) {
  int i = blockIdx.x * blockDim.x + threadIdx.x;
  if (i >= NUM_GRAPHS * DIM) return;
  int gi = i >> 7, f = i & 127;
  float acc = bc1[f];
  for (int k = 0; k < DIM; ++k) acc += pooled[gi * DIM + k] * C1[k * DIM + f];
  g1[i] = fmaxf(acc, 0.0f);
}

__global__ void mlp2(const float* __restrict__ g1,
                     const float* __restrict__ C2,
                     const float* __restrict__ bc2, float* __restrict__ out) {
  int i = blockIdx.x * blockDim.x + threadIdx.x;
  if (i >= NUM_GRAPHS * NUM_CLASSES) return;
  int gi = i / NUM_CLASSES, c = i % NUM_CLASSES;
  float acc = bc2[c];
  for (int k = 0; k < DIM; ++k) acc += g1[gi * DIM + k] * C2[k * NUM_CLASSES + c];
  out[i] = acc;
}

// ---------------- launch ----------------
extern "C" void kernel_launch(void* const* d_in, const int* in_sizes, int n_in,
                              void* d_out, int out_size, void* d_ws, size_t ws_size,
                              hipStream_t stream) {
  const float*     x     = (const float*)d_in[0];
  const long long* ei    = (const long long*)d_in[1];
  const long long* src   = ei;
  const long long* dst   = ei + N_EDGES;
  const long long* batch = (const long long*)d_in[2];
  const float* W1 = (const float*)d_in[3];  const float* b1  = (const float*)d_in[4];
  const float* W2 = (const float*)d_in[5];  const float* b2  = (const float*)d_in[6];
  const float* W3 = (const float*)d_in[7];  const float* b3  = (const float*)d_in[8];
  const float* C1 = (const float*)d_in[9];  const float* bc1 = (const float*)d_in[10];
  const float* C2 = (const float*)d_in[11]; const float* bc2 = (const float*)d_in[12];

  float* ws     = (float*)d_ws;
  float* dinv   = ws + OFF_DINV;
  float* norm   = ws + OFF_NORM;
  float* hA     = ws + OFF_HA;    // GEMM output
  float* hB     = ws + OFF_HB;    // aggregated activations
  float* pooled = ws + OFF_POOL;
  float* counts = ws + OFF_CNT;
  float* g1     = ws + OFF_G1;

  const int T = 256;
  // normalization coefficients (once, reused by all 3 layers)
  init_deg   <<<(N_NODES + T - 1) / T, T, 0, stream>>>(dinv);
  deg_scatter<<<(N_EDGES + T - 1) / T, T, 0, stream>>>(dst, dinv);
  deg_to_dinv<<<(N_NODES + T - 1) / T, T, 0, stream>>>(dinv);
  edge_norm  <<<(N_EDGES + T - 1) / T, T, 0, stream>>>(src, dst, dinv, norm);

  const int GEMM_GRID  = N_NODES / ROWS_PER_BLK;       // 625, exact
  const int EDGE_GRID  = (N_EDGES + 7) / 8;            // 8 waves/block
  const int NODE4_GRID = (N_NODES * (DIM / 4) + T - 1) / T;
  const int NF_GRID    = (N_NODES * DIM + T - 1) / T;

  // ---- layer 1: x -> hB ----
  gemm_xw       <<<GEMM_GRID, T, 0, stream>>>(x, W1, hA);
  self_loop_init<<<NODE4_GRID, T, 0, stream>>>(hA, dinv, hB);
  edge_scatter  <<<EDGE_GRID, T, 0, stream>>>(src, dst, norm, hA, hB);
  bias_act      <<<NF_GRID, T, 0, stream>>>(hB, b1, 1);
  // ---- layer 2: hB -> hB ----
  gemm_xw       <<<GEMM_GRID, T, 0, stream>>>(hB, W2, hA);
  self_loop_init<<<NODE4_GRID, T, 0, stream>>>(hA, dinv, hB);
  edge_scatter  <<<EDGE_GRID, T, 0, stream>>>(src, dst, norm, hA, hB);
  bias_act      <<<NF_GRID, T, 0, stream>>>(hB, b2, 1);
  // ---- layer 3 (no relu): hB -> hB ----
  gemm_xw       <<<GEMM_GRID, T, 0, stream>>>(hB, W3, hA);
  self_loop_init<<<NODE4_GRID, T, 0, stream>>>(hA, dinv, hB);
  edge_scatter  <<<EDGE_GRID, T, 0, stream>>>(src, dst, norm, hA, hB);
  bias_act      <<<NF_GRID, T, 0, stream>>>(hB, b3, 0);

  // ---- global mean pool ----
  zero_buf  <<<(NUM_GRAPHS * DIM + NUM_GRAPHS + T - 1) / T, T, 0, stream>>>(
      pooled, NUM_GRAPHS * DIM + NUM_GRAPHS);
  pool_count<<<(N_NODES + T - 1) / T, T, 0, stream>>>(batch, counts);
  pool_sum  <<<NF_GRID, T, 0, stream>>>(batch, hB, pooled);
  pool_div  <<<(NUM_GRAPHS * DIM + T - 1) / T, T, 0, stream>>>(pooled, counts);

  // ---- classifier head ----
  mlp1<<<(NUM_GRAPHS * DIM + T - 1) / T, T, 0, stream>>>(pooled, C1, bc1, g1);
  mlp2<<<(NUM_GRAPHS * NUM_CLASSES + T - 1) / T, T, 0, stream>>>(g1, C2, bc2,
                                                                 (float*)d_out);
}
